// DeepESN_54382875902079
// MI455X (gfx1250) — matrix operand order repaired and checked by reading, not compiled, not run on hardware
//
#include <hip/hip_runtime.h>
#include <math.h>

// Problem constants (match reference)
#define B_ 16
#define T_ 2048
#define U_ 512
#define I_ 128
#define OUTC_ (3 * U_)

typedef __attribute__((ext_vector_type(16))) _Float16 v16h;
typedef __attribute__((ext_vector_type(8)))  _Float16 v8h;
typedef __attribute__((ext_vector_type(8)))  float    v8f;

// ---------------------------------------------------------------------------
// tanh: prefer the gfx1250 v_tanh_f32 trans op; otherwise a 4-op approximation
// built from hardware exp2/rcp (error ~1e-6, far below the f16 WMMA noise).
// ---------------------------------------------------------------------------
__device__ __forceinline__ float esn_tanh(float x) {
#if defined(__has_builtin) && __has_builtin(__builtin_amdgcn_tanhf)
    return __builtin_amdgcn_tanhf(x);
#else
    float xx = fminf(fmaxf(x, -10.f), 10.f);
#if defined(__has_builtin) && __has_builtin(__builtin_amdgcn_exp2f) && __has_builtin(__builtin_amdgcn_rcpf)
    const float e = __builtin_amdgcn_exp2f(xx * 2.8853900817779268f); // 2*log2(e)
    const float r = __builtin_amdgcn_rcpf(e + 1.f);
#else
    const float e = __expf(2.f * xx);
    const float r = 1.f / (e + 1.f);
#endif
    return fmaf(-2.f, r, 1.f);
#endif
}

// ---------------------------------------------------------------------------
// Input projection: pre[(b*T+t)][n] = sum_k X[row][k] * W[k][n] + bias[n]
// WMMA f16 GEMM. Each wave owns one 16-col N-tile and four 16-row M-tiles.
// Grid: (512, 4) blocks x 256 threads -> 2048 M-tiles x 32 N-tiles.
// ---------------------------------------------------------------------------
__global__ __launch_bounds__(256) void esn_proj(const float* __restrict__ X, int ldx, int K,
                                                const float* __restrict__ W,
                                                const float* __restrict__ bias,
                                                void* __restrict__ preAny, int preF16)
{
    const int lane  = threadIdx.x & 31;
    const int wave  = threadIdx.x >> 5;       // 0..7
    const int hi    = lane >> 4;              // 0/1
    const int l16   = lane & 15;
    const int nbase = (blockIdx.y * 8 + wave) * 16;   // N-tile base (0..496)
    const int mbase = blockIdx.x * 64;                // 4 M-tiles of 16 rows

    v8f acc[4] = {};
    const int nchunks = K >> 5;
    for (int kc = 0; kc < nchunks; ++kc) {
        const int kb = kc * 32;
        // B fragment (32x16 f16): elem e -> W[(kb + 16*hi + e)*U + nbase + l16]
        v16h bf;
#pragma unroll
        for (int e = 0; e < 16; ++e)
            bf[e] = (_Float16)W[(size_t)(kb + 16 * hi + e) * U_ + nbase + l16];
#pragma unroll
        for (int m = 0; m < 4; ++m) {
            const float* xr = X + (size_t)(mbase + m * 16 + l16) * ldx + kb;
            // A fragment (16x32 f16): elem e -> K = 16*hi+e
            v16h af;
#pragma unroll
            for (int e = 0; e < 8; ++e) {
                af[e]     = (_Float16)xr[16 * hi + e];
                af[e + 8] = (_Float16)xr[16 * hi + 8 + e];
            }
            acc[m] = __builtin_amdgcn_wmma_f32_16x16x32_f16(
                false, af, false, bf, (short)0, acc[m], false, false);
        }
    }

    const float bv = bias[nbase + l16];
#pragma unroll
    for (int m = 0; m < 4; ++m) {
#pragma unroll
        for (int r = 0; r < 8; ++r) {
            const size_t idx = (size_t)(mbase + m * 16 + r + 8 * hi) * U_ + nbase + l16;
            const float v = acc[m][r] + bv;
            if (preF16) ((_Float16*)preAny)[idx] = (_Float16)v;
            else        ((float*)preAny)[idx]    = v;
        }
    }
}

// ---------------------------------------------------------------------------
// Sequential recurrence for one layer:
//   h_t = tanh(pre_t + h_{t-1} @ Wrec), written to out[:, :, colOff:colOff+512]
// One persistent 1024-thread workgroup (32 waves). Wave w owns N-tile w.
// Wrec is resident in VGPRs as 16 f16 B-fragments per wave; h is double-
// buffered in LDS as f16; one s_barrier per step is the only synchronization.
// pre_t is register double-buffered: step t issues step t+1's loads before the
// WMMA chain so global latency hides under the matrix ops.
// ---------------------------------------------------------------------------
__global__ __launch_bounds__(1024) void esn_recur(const void* __restrict__ preAny,
                                                  const float* __restrict__ Wrec,
                                                  float* __restrict__ out,
                                                  int colOff, int preF16)
{
    __shared__ _Float16 hbuf[2][B_ * U_];   // 2 x 16KB double buffer

    const int lane = threadIdx.x & 31;
    const int wave = threadIdx.x >> 5;      // 0..31 -> N-tile index
    const int hi   = lane >> 4;
    const int l16  = lane & 15;
    const int n    = wave * 16 + l16;       // output unit column

    // h_{-1} = 0
    for (int i = threadIdx.x; i < B_ * U_; i += 1024)
        hbuf[0][i] = (_Float16)0.f;

    // Resident Wrec B-fragments (f16), 16 chunks of K=32 for this wave's tile.
    v16h Bf[16];
#pragma unroll
    for (int kc = 0; kc < 16; ++kc) {
#pragma unroll
        for (int e = 0; e < 16; ++e)
            Bf[kc][e] = (_Float16)Wrec[(size_t)(kc * 32 + 16 * hi + e) * U_ + n];
    }
    __syncthreads();

    const float*    preF = (const float*)preAny;
    const _Float16* preH = (const _Float16*)preAny;

    // Preload pre for t=0
    v8f c;
#pragma unroll
    for (int r = 0; r < 8; ++r) {
        const size_t idx = ((size_t)(r + 8 * hi) * T_ + 0) * U_ + n;
        c[r] = preF16 ? (float)preH[idx] : preF[idx];
    }

    for (int t = 0; t < T_; ++t) {
        const _Float16* cur = hbuf[t & 1];
        _Float16*       nxt = hbuf[(t + 1) & 1];

        // Issue next step's pre loads now; they retire under the WMMA chain.
        v8f cn = {};
        if (t + 1 < T_) {
#pragma unroll
            for (int r = 0; r < 8; ++r) {
                const size_t idx = ((size_t)(r + 8 * hi) * T_ + (t + 1)) * U_ + n;
                cn[r] = preF16 ? (float)preH[idx] : preF[idx];
            }
        }

        // h_{t-1} @ Wrec : 16 WMMA ops, A from LDS, B resident in VGPRs
        const _Float16* arow = cur + (size_t)l16 * U_;
#pragma unroll
        for (int kc = 0; kc < 16; ++kc) {
            const v8h lo = *(const v8h*)(arow + kc * 32 + 16 * hi);
            const v8h hh = *(const v8h*)(arow + kc * 32 + 16 * hi + 8);
            v16h a = __builtin_shufflevector(lo, hh,
                0, 1, 2, 3, 4, 5, 6, 7, 8, 9, 10, 11, 12, 13, 14, 15);
            c = __builtin_amdgcn_wmma_f32_16x16x32_f16(
                false, a, false, Bf[kc], (short)0, c, false, false);
        }

        // h_t = tanh(c); write to output slice and to next LDS h buffer
#pragma unroll
        for (int r = 0; r < 8; ++r) {
            const int b = r + 8 * hi;
            const float hv = esn_tanh(c[r]);
            nxt[b * U_ + n] = (_Float16)hv;
            out[((size_t)b * T_ + t) * OUTC_ + colOff + n] = hv;
        }

        c = cn;
        __syncthreads();
    }
}

// ---------------------------------------------------------------------------
// Host-side launch: for each layer, projection GEMM then recurrence.
// d_in order: x, Win0, Wrec0, b0, Win1, Wrec1, b1, Win2, Wrec2, b2
// ---------------------------------------------------------------------------
extern "C" void kernel_launch(void* const* d_in, const int* in_sizes, int n_in,
                              void* d_out, int out_size, void* d_ws, size_t ws_size,
                              hipStream_t stream)
{
    (void)in_sizes; (void)n_in; (void)out_size;
    const float* x   = (const float*)d_in[0];
    float*       out = (float*)d_out;

    const size_t preElems = (size_t)B_ * T_ * U_;                    // 16.7M elems
    const int preF16 = (ws_size < preElems * sizeof(float)) ? 1 : 0; // f32 needs 64MB

    for (int l = 0; l < 3; ++l) {
        const float* Win  = (const float*)d_in[1 + 3 * l];
        const float* Wrec = (const float*)d_in[2 + 3 * l];
        const float* bias = (const float*)d_in[3 + 3 * l];
        const float* src  = (l == 0) ? x : (out + (size_t)U_ * (l - 1));
        const int    ldx  = (l == 0) ? I_ : OUTC_;
        const int    K    = (l == 0) ? I_ : U_;

        esn_proj<<<dim3(512, 4), 256, 0, stream>>>(src, ldx, K, Win, bias, d_ws, preF16);
        esn_recur<<<dim3(1), 1024, 0, stream>>>(d_ws, Wrec, out, U_ * l, preF16);
    }
}
